// SelectionGNN_73632919322944
// MI455X (gfx1250) — matrix operand order repaired and checked by reading, not compiled, not run on hardware
//
#include <hip/hip_runtime.h>

// ---------------------------------------------------------------------------
// Selection GNN on MI455X (gfx1250).
// Dominant cost: 6 dense graph shifts z <- S z (S: 8192x8192 fp32, 256MB).
// Strategy:
//   - if workspace allows (~150MB): convert S to bf16 once, stream bf16
//     (6 x 128MB HBM instead of 6 x 256MB)
//   - bf16 WMMA (v_wmma_f32_16x16x32_bf16), fp32 accumulation
//   - Z (RHS) staged into LDS with GLOBAL_LOAD_ASYNC_TO_LDS_B128, 4-buffer
//     depth-3 pipeline synchronized with s_wait_asynccnt + 1 barrier/k-step
//   - pipeline drain peeled out of the main loop so the hot loop has no
//     conditional branches (constant wait thresholds per peel stage)
// ---------------------------------------------------------------------------

typedef __attribute__((ext_vector_type(16))) __bf16 v16bf;
typedef __attribute__((ext_vector_type(8)))  float  v8f;

#define GNN_N   8192
#define GNN_N1  2048
#define GNN_N2  512
#define GNN_B   16
#define GNN_K   4
#define GNN_F1  8
#define GNN_F2  16
#define GNN_H1  256
#define GNN_H2  64
#define GNN_NBH 16
#define ZK      (128 * GNN_N)   // elements per Z tap buffer (bf16)

__device__ __forceinline__ void async_ld_b128(unsigned int lds_off,
                                              const void* gaddr) {
  // GLOBAL_LOAD_ASYNC_TO_LDS_B128: vdst = LDS byte address, vaddr = 64b addr.
  asm volatile("global_load_async_to_lds_b128 %0, %1, off"
               :
               : "v"(lds_off), "v"(gaddr)
               : "memory");
}

#define WAIT_ASYNC(n) asm volatile("s_wait_asynccnt %0" ::"i"(n) : "memory")

// ---------------------------------------------------------------------------
// WMMA shift kernel: Zout[c][n] = sum_m S[n][m] * Zin[c][m]
//   block = 256 threads = 8 waves; wave w owns node-tile m0..m0+15 and loops
//   all CH/16 channel groups (A fragment reused CG times per k-step).
// ---------------------------------------------------------------------------
template <int CH, bool ABF16>
__global__ __launch_bounds__(256) void shift_wmma(
    const void* __restrict__ Sv, const __bf16* __restrict__ Zin,
    __bf16* __restrict__ Zout) {
  constexpr int N = GNN_N;
  constexpr int CG = CH / 16;
  constexpr int LSTR = 40;                   // padded channel stride (bf16)
  constexpr int CHUNKS = CH * 4;             // 16B chunks per 32-wide slice
  constexpr int CPT = (CHUNKS + 255) / 256;  // chunks per thread
  constexpr int NSLICES = N / 32;
  constexpr int AREGS = ABF16 ? 2 : 4;

  __shared__ __bf16 ldsB[4][CH * LSTR];

  const int tid  = threadIdx.x;
  const int wave = tid >> 5;
  const int lane = tid & 31;
  const int half = lane >> 4;  // 0: lanes 0-15, 1: lanes 16-31
  const int l16  = lane & 15;
  const int m0   = (blockIdx.x * 8 + wave) * 16;
  const size_t rowoff = (size_t)(m0 + l16) * N;

  v8f acc[CG];
#pragma unroll
  for (int cg = 0; cg < CG; ++cg) acc[cg] = (v8f)0.0f;

  int sc[CPT], sseg[CPT];
#pragma unroll
  for (int i = 0; i < CPT; ++i) {
    int id = tid + i * 256;
    sc[i] = id >> 2;          // channel
    sseg[i] = (id & 3) * 8;   // bf16 offset within 32-wide slice
  }

  // issue async loads for k-slice `sl` into LDS buffer sl%4
  auto issueZ = [&](int sl) {
    const int buf = sl & 3;
#pragma unroll
    for (int i = 0; i < CPT; ++i)
      if (tid + i * 256 < CHUNKS) {
        unsigned int loff =
            (unsigned int)(uintptr_t)&ldsB[buf][sc[i] * LSTR + sseg[i]];
        const void* g =
            (const void*)(Zin + (size_t)sc[i] * N + sl * 32 + sseg[i]);
        async_ld_b128(loff, g);
      }
  };

  auto loadA = [&](uint4* r, int k0) {
    if constexpr (ABF16) {
      const __bf16* s = (const __bf16*)Sv + rowoff;
      r[0] = *reinterpret_cast<const uint4*>(s + k0 + half * 8);
      r[1] = *reinterpret_cast<const uint4*>(s + k0 + 16 + half * 8);
    } else {
      const float* s = (const float*)Sv + rowoff;
      const uint4* p0 = reinterpret_cast<const uint4*>(s + k0 + half * 8);
      const uint4* p1 = reinterpret_cast<const uint4*>(s + k0 + 16 + half * 8);
      r[0] = p0[0];
      r[1] = p0[1];
      r[2] = p1[0];
      r[3] = p1[1];
    }
  };

  // WMMAs for k-slice j using A regs `a` and LDS buffer j%4
  auto compute = [&](int j, const uint4* a) {
    // A fragment 16x32 bf16:
    // lanes 0-15: K = k0+0..7, k0+16..23 ; lanes 16-31: +8
    union {
      uint4 u[2];
      __bf16 h[16];
      v16bf v;
    } ua;
    if constexpr (ABF16) {
      ua.u[0] = a[0];
      ua.u[1] = a[1];
    } else {
      union {
        uint4 u[4];
        float f[16];
      } uf;
#pragma unroll
      for (int i = 0; i < 4; ++i) uf.u[i] = a[i];
#pragma unroll
      for (int i = 0; i < 16; ++i) ua.h[i] = (__bf16)uf.f[i];
    }
    const int buf = j & 3;
#pragma unroll
    for (int cg = 0; cg < CG; ++cg) {
      // B fragment 32x16: lane = column (channel), same K order as A:
      // lanes 0-15: K = 0..7, 16..23 ; lanes 16-31: +8.
      const __bf16* pb = &ldsB[buf][(cg * 16 + l16) * LSTR + half * 8];
      union {
        uint4 u[2];
        v16bf v;
      } ub;
      ub.u[0] = *reinterpret_cast<const uint4*>(pb);
      ub.u[1] = *reinterpret_cast<const uint4*>(pb + 16);
      acc[cg] = __builtin_amdgcn_wmma_f32_16x16x32_bf16(
          false, ua.v, false, ub.v, (short)0, acc[cg], false, false);
    }
  };

  // prologue: slices 0..2 in flight, A regs for slice 0
  issueZ(0);
  issueZ(1);
  issueZ(2);
  uint4 a_cur[AREGS];
  loadA(a_cur, 0);

  // ---- main loop: branch-free steady state (252 iterations) ----
#pragma unroll 2
  for (int j = 0; j < NSLICES - 4; ++j) {
    WAIT_ASYNC(2 * CPT);  // retire slice j; j+1, j+2 stay in flight
    __syncthreads();      // implicit s_wait_loadcnt 0 retires PREVIOUS A loads
    issueZ(j + 3);        // refill buffer freed at iteration j-1
    uint4 a_nxt[AREGS];   // A prefetch AFTER barrier: flies through WMMAs
    loadA(a_nxt, (j + 1) * 32);
    compute(j, a_cur);
#pragma unroll
    for (int i = 0; i < AREGS; ++i) a_cur[i] = a_nxt[i];
  }

  // ---- pipeline drain: constant thresholds, no branches ----
  {  // j = NSLICES-4: last issue (slice NSLICES-1)
    constexpr int j = NSLICES - 4;
    WAIT_ASYNC(2 * CPT);
    __syncthreads();
    issueZ(j + 3);
    uint4 a_nxt[AREGS];
    loadA(a_nxt, (j + 1) * 32);
    compute(j, a_cur);
#pragma unroll
    for (int i = 0; i < AREGS; ++i) a_cur[i] = a_nxt[i];
  }
  {  // j = NSLICES-3: slices j+1, j+2 still outstanding
    constexpr int j = NSLICES - 3;
    WAIT_ASYNC(2 * CPT);
    __syncthreads();
    uint4 a_nxt[AREGS];
    loadA(a_nxt, (j + 1) * 32);
    compute(j, a_cur);
#pragma unroll
    for (int i = 0; i < AREGS; ++i) a_cur[i] = a_nxt[i];
  }
  {  // j = NSLICES-2: slice j+1 still outstanding
    constexpr int j = NSLICES - 2;
    WAIT_ASYNC(CPT);
    __syncthreads();
    uint4 a_nxt[AREGS];
    loadA(a_nxt, (j + 1) * 32);
    compute(j, a_cur);
#pragma unroll
    for (int i = 0; i < AREGS; ++i) a_cur[i] = a_nxt[i];
  }
  {  // j = NSLICES-1: drain fully
    constexpr int j = NSLICES - 1;
    WAIT_ASYNC(0);
    __syncthreads();
    compute(j, a_cur);
  }

  // D layout: VGPR r -> node m0 + r + 8*half, channel cg*16 + l16.
#pragma unroll
  for (int cg = 0; cg < CG; ++cg) {
    const int c = cg * 16 + l16;
    union {
      __bf16 h[8];
      uint4 u;
    } uo;
#pragma unroll
    for (int r = 0; r < 8; ++r) uo.h[r] = (__bf16)acc[cg][r];
    *reinterpret_cast<uint4*>(Zout + (size_t)c * N + m0 + half * 8) = uo.u;
  }
}

// ---------------------------------------------------------------------------
// Elementwise / gather kernels
// ---------------------------------------------------------------------------
__global__ void cvt_S_kernel(const float* __restrict__ S,
                             __bf16* __restrict__ Sb) {
  size_t idx = ((size_t)blockIdx.x * 256 + threadIdx.x) * 8;
  const float4* p = reinterpret_cast<const float4*>(S + idx);
  float4 a = p[0], b = p[1];
  union {
    __bf16 h[8];
    uint4 u;
  } uo;
  uo.h[0] = (__bf16)a.x;
  uo.h[1] = (__bf16)a.y;
  uo.h[2] = (__bf16)a.z;
  uo.h[3] = (__bf16)a.w;
  uo.h[4] = (__bf16)b.x;
  uo.h[5] = (__bf16)b.y;
  uo.h[6] = (__bf16)b.z;
  uo.h[7] = (__bf16)b.w;
  *reinterpret_cast<uint4*>(Sb + idx) = uo.u;
}

__global__ void cvt_x_kernel(const float* __restrict__ x,
                             __bf16* __restrict__ Z0) {
  int idx = blockIdx.x * 256 + threadIdx.x;  // 16*8192
  Z0[idx] = (__bf16)x[idx];
}

__global__ void zfill_kernel(unsigned int* __restrict__ p) {
  int idx = blockIdx.x * 256 + threadIdx.x;
  p[idx] = 0u;
}

// y1[b][g][n] = relu(sum_k h1[g][k][0] * Z_k[b][n] + bf1[g])
__global__ void combine1_kernel(const __bf16* __restrict__ Z,
                                const float* __restrict__ h1,
                                const float* __restrict__ bf1,
                                float* __restrict__ y1full) {
  int idx = blockIdx.x * 256 + threadIdx.x;  // B*F1*N
  int n = idx & (GNN_N - 1);
  int g = (idx >> 13) & (GNN_F1 - 1);
  int b = idx >> 16;
  float acc = bf1[g];
#pragma unroll
  for (int k = 0; k < GNN_K; ++k)
    acc += h1[g * GNN_K + k] * (float)Z[(size_t)k * ZK + (size_t)b * GNN_N + n];
  y1full[idx] = fmaxf(acc, 0.0f);
}

// pooled layer-1 output becomes layer-2 z0 (bf16, zero-padded to N)
__global__ void pool1_kernel(const float* __restrict__ y1full,
                             const int* __restrict__ nbh1,
                             __bf16* __restrict__ Z0) {
  int idx = blockIdx.x * 256 + threadIdx.x;  // B*F1*N1
  int j = idx & (GNN_N1 - 1);
  int g = (idx >> 11) & (GNN_F1 - 1);
  int b = idx >> 14;
  const float* row = y1full + (size_t)(b * GNN_F1 + g) * GNN_N;
  const int* nb = nbh1 + j * GNN_NBH;
  float m = row[nb[0]];
#pragma unroll
  for (int t = 1; t < GNN_NBH; ++t) m = fmaxf(m, row[nb[t]]);
  Z0[(size_t)(b * GNN_F1 + g) * GNN_N + j] = (__bf16)m;
}

// y2[b][g][n<N1] = relu(sum_{k,f} h2[g][k][f] * Z_k[b*8+f][n] + bf2[g])
__global__ void combine2_kernel(const __bf16* __restrict__ Z,
                                const float* __restrict__ h2,
                                const float* __restrict__ bf2,
                                float* __restrict__ y2full) {
  int idx = blockIdx.x * 256 + threadIdx.x;  // B*F2*N1
  int n = idx & (GNN_N1 - 1);
  int g = (idx >> 11) & (GNN_F2 - 1);
  int b = idx >> 15;
  float acc = bf2[g];
#pragma unroll
  for (int k = 0; k < GNN_K; ++k)
#pragma unroll
    for (int f = 0; f < GNN_F1; ++f)
      acc += h2[g * (GNN_K * GNN_F1) + k * GNN_F1 + f] *
             (float)Z[(size_t)k * ZK + (size_t)(b * GNN_F1 + f) * GNN_N + n];
  y2full[idx] = fmaxf(acc, 0.0f);
}

__global__ void pool2_kernel(const float* __restrict__ y2full,
                             const int* __restrict__ nbh2,
                             float* __restrict__ flat) {
  int idx = blockIdx.x * 256 + threadIdx.x;  // B*F2*N2
  int j = idx & (GNN_N2 - 1);
  int g = (idx >> 9) & (GNN_F2 - 1);
  int b = idx >> 13;
  const float* row = y2full + (size_t)(b * GNN_F2 + g) * GNN_N1;
  const int* nb = nbh2 + j * GNN_NBH;
  float m = row[nb[0]];
#pragma unroll
  for (int t = 1; t < GNN_NBH; ++t) m = fmaxf(m, row[nb[t]]);
  flat[(size_t)b * (GNN_F2 * GNN_N2) + g * GNN_N2 + j] = m;
}

// hid partial sums: grid = B*8 blocks, 256 threads (one per H1 column),
// each block handles a 1024-wide K chunk. Deterministic (no atomics).
__global__ void mlp1_kernel(const float* __restrict__ flat,
                            const float* __restrict__ W1,
                            float* __restrict__ hid_part) {
  int b = blockIdx.x >> 3;
  int p = blockIdx.x & 7;
  int h = threadIdx.x;
  int k0 = p * 1024;
  const float* xr = flat + (size_t)b * (GNN_F2 * GNN_N2) + k0;
  const float* w = W1 + (size_t)k0 * GNN_H1 + h;
  float acc = 0.0f;
#pragma unroll 8
  for (int k = 0; k < 1024; ++k) acc += xr[k] * w[(size_t)k * GNN_H1];
  hid_part[(size_t)(p * GNN_B + b) * GNN_H1 + h] = acc;
}

__global__ void mlp2_kernel(const float* __restrict__ hid_part,
                            const float* __restrict__ bm1,
                            const float* __restrict__ W2,
                            const float* __restrict__ bm2,
                            float* __restrict__ out) {
  int idx = blockIdx.x * 256 + threadIdx.x;  // B*H2 = 1024
  int o = idx & (GNN_H2 - 1);
  int b = idx >> 6;
  float acc = bm2[o];
  for (int h = 0; h < GNN_H1; ++h) {
    float v = bm1[h];
#pragma unroll
    for (int p = 0; p < 8; ++p)
      v += hid_part[(size_t)(p * GNN_B + b) * GNN_H1 + h];
    acc += fmaxf(v, 0.0f) * W2[(size_t)h * GNN_H2 + o];
  }
  out[idx] = acc;
}

// ---------------------------------------------------------------------------
extern "C" void kernel_launch(void* const* d_in, const int* in_sizes, int n_in,
                              void* d_out, int out_size, void* d_ws,
                              size_t ws_size, hipStream_t stream) {
  const float* x   = (const float*)d_in[0];
  const float* S   = (const float*)d_in[1];
  const float* h1  = (const float*)d_in[2];
  const float* bf1 = (const float*)d_in[3];
  const float* h2  = (const float*)d_in[4];
  const float* bf2 = (const float*)d_in[5];
  const float* W1  = (const float*)d_in[6];
  const float* bm1 = (const float*)d_in[7];
  const float* W2  = (const float*)d_in[8];
  const float* bm2 = (const float*)d_in[9];
  const int* nbh1  = (const int*)d_in[10];
  const int* nbh2  = (const int*)d_in[11];
  float* out = (float*)d_out;

  const size_t sb_bytes = (size_t)GNN_N * GNN_N * sizeof(__bf16);  // 128 MB
  const size_t rest_bytes =
      (size_t)4 * ZK * sizeof(__bf16) +
      ((size_t)GNN_B * GNN_F1 * GNN_N + (size_t)GNN_B * GNN_F2 * GNN_N1 +
       (size_t)GNN_B * GNN_F2 * GNN_N2 + (size_t)8 * GNN_B * GNN_H1) *
          sizeof(float);
  const bool use_sb = ws_size >= sb_bytes + rest_bytes;

  char* w = (char*)d_ws;
  __bf16* Sb = nullptr;
  if (use_sb) {
    Sb = (__bf16*)w;
    w += sb_bytes;
  }
  __bf16* Z = (__bf16*)w;  // 4 * ZK bf16
  float* y1full = (float*)(w + (size_t)4 * ZK * sizeof(__bf16));
  float* y2full = y1full + (size_t)GNN_B * GNN_F1 * GNN_N;
  float* flat   = y2full + (size_t)GNN_B * GNN_F2 * GNN_N1;
  float* hidp   = flat + (size_t)GNN_B * GNN_F2 * GNN_N2;

  __bf16* Z0 = Z;
  __bf16* Z1 = Z + (size_t)1 * ZK;
  __bf16* Z2 = Z + (size_t)2 * ZK;
  __bf16* Z3 = Z + (size_t)3 * ZK;

  if (use_sb)
    cvt_S_kernel<<<(GNN_N * GNN_N / 8) / 256, 256, 0, stream>>>(S, Sb);

  // ---- layer 1: z0 = x (bf16), 3 shifts over 16 channels ----
  cvt_x_kernel<<<(GNN_B * GNN_N) / 256, 256, 0, stream>>>(x, Z0);
  if (use_sb) {
    shift_wmma<16, true><<<GNN_N / 128, 256, 0, stream>>>(Sb, Z0, Z1);
    shift_wmma<16, true><<<GNN_N / 128, 256, 0, stream>>>(Sb, Z1, Z2);
    shift_wmma<16, true><<<GNN_N / 128, 256, 0, stream>>>(Sb, Z2, Z3);
  } else {
    shift_wmma<16, false><<<GNN_N / 128, 256, 0, stream>>>(S, Z0, Z1);
    shift_wmma<16, false><<<GNN_N / 128, 256, 0, stream>>>(S, Z1, Z2);
    shift_wmma<16, false><<<GNN_N / 128, 256, 0, stream>>>(S, Z2, Z3);
  }
  combine1_kernel<<<(GNN_B * GNN_F1 * GNN_N) / 256, 256, 0, stream>>>(
      Z, h1, bf1, y1full);

  // ---- pool 1 -> layer-2 z0 (zero padded 128 x 8192 bf16) ----
  zfill_kernel<<<(ZK / 2) / 256, 256, 0, stream>>>((unsigned int*)Z0);
  pool1_kernel<<<(GNN_B * GNN_F1 * GNN_N1) / 256, 256, 0, stream>>>(
      y1full, nbh1, Z0);

  // ---- layer 2: 3 shifts over 128 channels ----
  if (use_sb) {
    shift_wmma<128, true><<<GNN_N / 128, 256, 0, stream>>>(Sb, Z0, Z1);
    shift_wmma<128, true><<<GNN_N / 128, 256, 0, stream>>>(Sb, Z1, Z2);
    shift_wmma<128, true><<<GNN_N / 128, 256, 0, stream>>>(Sb, Z2, Z3);
  } else {
    shift_wmma<128, false><<<GNN_N / 128, 256, 0, stream>>>(S, Z0, Z1);
    shift_wmma<128, false><<<GNN_N / 128, 256, 0, stream>>>(S, Z1, Z2);
    shift_wmma<128, false><<<GNN_N / 128, 256, 0, stream>>>(S, Z2, Z3);
  }
  combine2_kernel<<<(GNN_B * GNN_F2 * GNN_N1) / 256, 256, 0, stream>>>(
      Z, h2, bf2, y2full);

  // ---- pool 2 -> flat [B, F2*N2] ----
  pool2_kernel<<<(GNN_B * GNN_F2 * GNN_N2) / 256, 256, 0, stream>>>(
      y2full, nbh2, flat);

  // ---- MLP ----
  mlp1_kernel<<<GNN_B * 8, GNN_H1, 0, stream>>>(flat, W1, hidp);
  mlp2_kernel<<<(GNN_B * GNN_H2) / 256, 256, 0, stream>>>(hidp, bm1, W2, bm2,
                                                          out);
}
